// LambdaConv_55903294324769
// MI455X (gfx1250) — matrix operand order repaired and checked by reading, not compiled, not run on hardware
//
#include <hip/hip_runtime.h>
#include <hip/hip_bf16.h>

// ---------------- problem constants ----------------
#define Bn      16
#define CIN     512
#define Ln      2048
#define HEADSn  4
#define Kn      16
#define VVn     128
#define KSn     3
#define NCH     208          // 64 q + 16 k + 128 v channels
#define MTILES  13           // NCH/16
#define KTOT    1536         // CIN*KS
#define NCHUNK  48           // KTOT/32
#define EPSf    1e-5f

typedef __attribute__((ext_vector_type(16))) __bf16 v16bf;
typedef __attribute__((ext_vector_type(8)))  float  v8f;

// 16-bit A-matrix 16x32 K mapping (ISA 7.12.2): lane = m + 16*hi holds
// K = hi*8+i (i<8) and K = 16 + hi*8 + (i-8) (i>=8)
__device__ __forceinline__ int kmap(int hi, int i) {
    return (i < 8) ? (hi * 8 + i) : (16 + hi * 8 + (i - 8));
}

// ---------------- K0a: fold BN into weights, swizzle to A-fragment layout ----
// Aw layout: [MTILES][NCHUNK][32 lanes][16 bf16]  (each lane loads 32B)
__global__ void k_prep_weights(const float* __restrict__ Wq,
                               const float* __restrict__ Wk,
                               const float* __restrict__ Wv,
                               const float* __restrict__ qg, const float* __restrict__ qb,
                               const float* __restrict__ qm, const float* __restrict__ qvv,
                               const float* __restrict__ vg, const float* __restrict__ vb,
                               const float* __restrict__ vm, const float* __restrict__ vvv,
                               __bf16* __restrict__ Aw, float* __restrict__ bias) {
    int t = blockIdx.x * 256 + threadIdx.x;
    if (blockIdx.x == 0 && threadIdx.x < NCH) {
        int co = threadIdx.x; float bb;
        if (co < 64)      { float sc = qg[co] * rsqrtf(qvv[co] + EPSf); bb = qb[co] - qm[co] * sc; }
        else if (co < 80) { bb = 0.f; }
        else              { int c = co - 80; float sc = vg[c] * rsqrtf(vvv[c] + EPSf); bb = vb[c] - vm[c] * sc; }
        bias[co] = bb;
    }
    if (t >= MTILES * NCHUNK * 32) return;
    int lane = t & 31;
    int j    = (t >> 5) % NCHUNK;
    int mt   = t / (NCHUNK * 32);
    int m = lane & 15, hi = lane >> 4;
    int co = mt * 16 + m;
    float scale; const float* W; int col;
    if (co < 64)      { scale = qg[co] * rsqrtf(qvv[co] + EPSf); W = Wq; col = co; }
    else if (co < 80) { scale = 1.f;                             W = Wk; col = co - 64; }
    else              { int c = co - 80; scale = vg[c] * rsqrtf(vvv[c] + EPSf); W = Wv; col = c; }
    v16bf o;
    for (int i = 0; i < 16; ++i) {
        int kk = 32 * j + 16 * hi + i;     // K order: kk = t*512 + ci
        int tt = kk >> 9, ci = kk & 511;
        o[i] = (__bf16)(W[(size_t)col * CIN * KSn + (size_t)ci * KSn + tt] * scale);
    }
    *(v16bf*)&Aw[(((size_t)mt * NCHUNK + j) * 32 + lane) * 16] = o;
}

// ---------------- K0b: x -> bf16 transpose with halo: xT[b][n+1][ci] --------
__global__ void k_transpose_x(const float* __restrict__ x, __bf16* __restrict__ xT) {
    __shared__ float tile[16][17];
    int n0 = blockIdx.x * 16, c0 = blockIdx.y * 16, b = blockIdx.z;
    int tn = threadIdx.x & 15, tc = threadIdx.x >> 4;
    tile[tc][tn] = x[((size_t)b * CIN + c0 + tc) * Ln + n0 + tn];
    __syncthreads();
    int wn = threadIdx.x >> 4, wc = threadIdx.x & 15;
    xT[((size_t)b * (Ln + 2) + (n0 + wn + 1)) * CIN + c0 + wc] = (__bf16)tile[wc][wn];
}

__global__ void k_zero_halo(__bf16* __restrict__ xT) {
    int t = blockIdx.x * 256 + threadIdx.x;
    if (t >= Bn * 2 * CIN) return;
    int ci = t & 511, r = (t >> 9) & 1, b = t >> 10;
    xT[((size_t)b * (Ln + 2) + (r ? (Ln + 1) : 0)) * CIN + ci] = (__bf16)0.f;
}

// ---------------- K1: fused conv-as-GEMM via WMMA bf16 ----------------------
// Block = 8 waves sharing one 16-channel tile (mt): A tile (48 KB) staged into
// LDS once via gfx1250 async copy; each wave computes 16 ch x 64 positions.
__global__ void __launch_bounds__(256) k_conv_wmma(const __bf16* __restrict__ xT,
                                                   const __bf16* __restrict__ Aw,
                                                   const float* __restrict__ bias,
                                                   float* __restrict__ qkv) {
    __shared__ __bf16 Alds[NCHUNK * 512];              // 48 chunks x 32 lanes x 16 bf16 = 48 KB
    int w = threadIdx.x >> 5, lane = threadIdx.x & 31;
    int bi  = blockIdx.x;                              // 0..831
    int b   = bi / (MTILES * 4);
    int rem = bi % (MTILES * 4);
    int mt  = rem >> 2;
    int n0  = (rem & 3) * 512 + w * 64;
    int m = lane & 15, hi = lane >> 4;

    // ---- async-stage weight tile for this mt into LDS (ASYNCcnt path) ----
    {
        const __bf16* srcA = Aw + (size_t)mt * NCHUNK * 32 * 16;   // 48 KB
        unsigned ldsBase = (unsigned)(uintptr_t)(&Alds[0]);
        for (int it = 0; it < 12; ++it) {                          // 256 thr * 16 B * 12 = 48 KB
            unsigned byteOff = (unsigned)(threadIdx.x + it * 256) * 16u;
            unsigned ldsOff  = ldsBase + byteOff;
            asm volatile("global_load_async_to_lds_b128 %0, %1, %2"
                         :: "v"(ldsOff), "v"(byteOff), "s"(srcA) : "memory");
        }
        asm volatile("s_wait_asynccnt 0x0" ::: "memory");
        __syncthreads();
    }

    const __bf16* xbase = &xT[((size_t)b * (Ln + 2)) * CIN];
    v8f acc0 = {}, acc1 = {}, acc2 = {}, acc3 = {};
    for (int j = 0; j < NCHUNK; ++j) {
        int kk0 = 32 * j, tt = kk0 >> 9, ci0 = kk0 & 511;
        v16bf a = *(const v16bf*)&Alds[((size_t)j * 32 + lane) * 16];
        const __bf16* base = xbase + (size_t)ci0 + 16 * hi + (size_t)tt * CIN;
        v16bf b0 = *(const v16bf*)(base + (size_t)(n0 + 0  + m) * CIN);
        v16bf b1 = *(const v16bf*)(base + (size_t)(n0 + 16 + m) * CIN);
        v16bf b2 = *(const v16bf*)(base + (size_t)(n0 + 32 + m) * CIN);
        v16bf b3 = *(const v16bf*)(base + (size_t)(n0 + 48 + m) * CIN);
        // keep the activation stream ahead of the WMMA pipe (global_prefetch_b8)
        if (j + 1 < NCHUNK) {
            int kk1 = 32 * (j + 1), tt1 = kk1 >> 9, ci1 = kk1 & 511;
            __builtin_prefetch(xbase + (size_t)ci1 + 16 * hi + (size_t)(tt1 + n0 + m) * CIN, 0, 3);
        }
        acc0 = __builtin_amdgcn_wmma_f32_16x16x32_bf16(false, a, false, b0, (short)0, acc0, false, false);
        acc1 = __builtin_amdgcn_wmma_f32_16x16x32_bf16(false, a, false, b1, (short)0, acc1, false, false);
        acc2 = __builtin_amdgcn_wmma_f32_16x16x32_bf16(false, a, false, b2, (short)0, acc2, false, false);
        acc3 = __builtin_amdgcn_wmma_f32_16x16x32_bf16(false, a, false, b3, (short)0, acc3, false, false);
    }
    v8f accs[4] = {acc0, acc1, acc2, acc3};
    for (int s = 0; s < 4; ++s)
        for (int r = 0; r < 8; ++r) {
            int co = mt * 16 + r + 8 * hi;
            int n  = n0 + 16 * s + m;
            qkv[((size_t)b * NCH + co) * Ln + n] = accs[s][r] + bias[co];
        }
}

// ---------------- K2a: softmax over positions, emit bf16 A-fragments --------
// smA layout: [B][64 chunks][32 lanes][16 bf16]
__global__ void k_softmax(const float* __restrict__ qkv, __bf16* __restrict__ smA) {
    int b = blockIdx.x >> 4, k = blockIdx.x & 15;
    const float* row = qkv + ((size_t)b * NCH + 64 + k) * Ln;
    __shared__ float red[256];
    float mx = -3.0e38f;
    for (int n = threadIdx.x; n < Ln; n += 256) mx = fmaxf(mx, row[n]);
    red[threadIdx.x] = mx; __syncthreads();
    for (int s = 128; s > 0; s >>= 1) {
        if (threadIdx.x < s) red[threadIdx.x] = fmaxf(red[threadIdx.x], red[threadIdx.x + s]);
        __syncthreads();
    }
    mx = red[0]; __syncthreads();
    float sum = 0.f;
    for (int n = threadIdx.x; n < Ln; n += 256) sum += __expf(row[n] - mx);
    red[threadIdx.x] = sum; __syncthreads();
    for (int s = 128; s > 0; s >>= 1) {
        if (threadIdx.x < s) red[threadIdx.x] += red[threadIdx.x + s];
        __syncthreads();
    }
    float rs = 1.f / red[0];
    if (threadIdx.x < 64) {
        int j = threadIdx.x;
        for (int hi = 0; hi < 2; ++hi) {
            v16bf o;
            for (int i = 0; i < 16; ++i) {
                int n = 32 * j + 16 * hi + i;
                o[i] = (__bf16)(__expf(row[n] - mx) * rs);
            }
            *(v16bf*)&smA[(((size_t)b * 64 + j) * 32 + (k + 16 * hi)) * 16] = o;
        }
    }
}

// ---------------- K2b: lambda_c[k][v] = sum_n sm[k][n]*v[v][n] via WMMA -----
__global__ void k_lambda(const __bf16* __restrict__ smA,
                         const float* __restrict__ qkv,
                         float* __restrict__ lam) {
    int b = blockIdx.x;
    int w = threadIdx.x >> 5, lane = threadIdx.x & 31;
    int m = lane & 15, hi = lane >> 4;
    int v0 = w * 16;
    v8f acc = {};
    for (int j = 0; j < 64; ++j) {
        v16bf a = *(const v16bf*)&smA[(((size_t)b * 64 + j) * 32 + lane) * 16];
        const float* vp = qkv + ((size_t)b * NCH + 80 + v0 + m) * Ln + 32 * j + 16 * hi;
        v16bf bb;
        for (int i = 0; i < 16; ++i) bb[i] = (__bf16)vp[i];
        acc = __builtin_amdgcn_wmma_f32_16x16x32_bf16(false, a, false, bb, (short)0, acc, false, false);
    }
    for (int r = 0; r < 8; ++r)
        lam[((size_t)b * 16 + (r + 8 * hi)) * VVn + v0 + m] = acc[r];
}

// ---------------- K2c: qT bf16 [b][n][64] + s[b][h][n] = sum_k q*emb --------
__global__ void k_qprep(const float* __restrict__ qkv, const float* __restrict__ emb,
                        __bf16* __restrict__ qT, float* __restrict__ sArr) {
    __shared__ float tile[64][17];
    int n0 = blockIdx.x * 16, b = blockIdx.y;
    int ch = threadIdx.x >> 2, ns = (threadIdx.x & 3) * 4;
    for (int i = 0; i < 4; ++i)
        tile[ch][ns + i] = qkv[((size_t)b * NCH + ch) * Ln + n0 + ns + i];
    __syncthreads();
    int nl = threadIdx.x >> 4, cg = (threadIdx.x & 15) * 4;
    for (int i = 0; i < 4; ++i)
        qT[((size_t)b * Ln + n0 + nl) * 64 + cg + i] = (__bf16)tile[cg + i][nl];
    if (threadIdx.x < 64) {
        int h = threadIdx.x >> 4, nn = threadIdx.x & 15;
        float s = 0.f;
        for (int k2 = 0; k2 < Kn; ++k2) s += tile[h * 16 + k2][nn] * emb[k2];
        sArr[((size_t)b * HEADSn + h) * Ln + n0 + nn] = s;
    }
}

// ---------------- K3: out = lambda^T x q (WMMA) + s .* v --------------------
__global__ void k_output(const __bf16* __restrict__ qT,
                         const float* __restrict__ lam,
                         const float* __restrict__ sArr,
                         const float* __restrict__ qkv,
                         float* __restrict__ out) {
    int wave = (int)((blockIdx.x * (size_t)blockDim.x + threadIdx.x) >> 5);
    int lane = threadIdx.x & 31;
    int vt = wave & 7, h = (wave >> 3) & 3, b = wave >> 5;
    int m = lane & 15, hi = lane >> 4;
    // A = lambda^T fragment (16 v x K=32, zero-padded for k>=16), loop-invariant
    v16bf a;
    for (int i = 0; i < 16; ++i) {
        int k = 16 * hi + i;
        a[i] = (k < Kn) ? (__bf16)lam[((size_t)b * 16 + k) * VVn + vt * 16 + m] : (__bf16)0.f;
    }
    for (int nt = 0; nt < Ln / 16; ++nt) {
        int n = nt * 16 + m;
        v16bf bb;
        if (hi == 0) {
            bb = *(const v16bf*)&qT[((size_t)b * Ln + n) * 64 + h * 16];
        } else {
            for (int i = 0; i < 16; ++i) bb[i] = (__bf16)0.f;   // k>=16 pad
        }
        v8f acc = {};
        acc = __builtin_amdgcn_wmma_f32_16x16x32_bf16(false, a, false, bb, (short)0, acc, false, false);
        float s = sArr[((size_t)b * HEADSn + h) * Ln + n];
        for (int r = 0; r < 8; ++r) {
            int vv = vt * 16 + r + 8 * hi;
            float vval = qkv[((size_t)b * NCH + 80 + vv) * Ln + n];
            out[((size_t)b * (HEADSn * VVn) + h * VVn + vv) * Ln + n] = acc[r] + s * vval;
        }
    }
}

// ---------------- launch ----------------------------------------------------
extern "C" void kernel_launch(void* const* d_in, const int* in_sizes, int n_in,
                              void* d_out, int out_size, void* d_ws, size_t ws_size,
                              hipStream_t stream) {
    const float* x   = (const float*)d_in[0];
    const float* Wq  = (const float*)d_in[1];
    const float* qg  = (const float*)d_in[2];
    const float* qb  = (const float*)d_in[3];
    const float* qm  = (const float*)d_in[4];
    const float* qv  = (const float*)d_in[5];
    const float* Wk  = (const float*)d_in[6];
    const float* Wv  = (const float*)d_in[7];
    const float* vg  = (const float*)d_in[8];
    const float* vb  = (const float*)d_in[9];
    const float* vm  = (const float*)d_in[10];
    const float* vvv = (const float*)d_in[11];
    const float* emb = (const float*)d_in[12];
    float* out = (float*)d_out;

    char* ws = (char*)d_ws;
    size_t o = 0;
    auto carve = [&](size_t bytes) { size_t r = o; o = (o + bytes + 255) & ~(size_t)255; return r; };
    __bf16* xT   = (__bf16*)(ws + carve((size_t)Bn * (Ln + 2) * CIN * 2));          // 33.6 MB
    __bf16* Aw   = (__bf16*)(ws + carve((size_t)MTILES * NCHUNK * 32 * 16 * 2));    // 0.64 MB
    float*  bias = (float*) (ws + carve((size_t)NCH * 4));
    float*  qkv  = (float*) (ws + carve((size_t)Bn * NCH * Ln * 4));                // 27.3 MB
    __bf16* smA  = (__bf16*)(ws + carve((size_t)Bn * 64 * 32 * 16 * 2));            // 1.0 MB
    float*  lam  = (float*) (ws + carve((size_t)Bn * Kn * VVn * 4));                // 0.13 MB
    __bf16* qT   = (__bf16*)(ws + carve((size_t)Bn * Ln * 64 * 2));                 // 4.2 MB
    float*  sArr = (float*) (ws + carve((size_t)Bn * HEADSn * Ln * 4));             // 0.52 MB
    (void)ws_size; (void)in_sizes; (void)n_in; (void)out_size;

    k_prep_weights<<<(MTILES * NCHUNK * 32 + 255) / 256, 256, 0, stream>>>(
        Wq, Wk, Wv, qg, qb, qm, qv, vg, vb, vm, vvv, Aw, bias);
    k_transpose_x<<<dim3(Ln / 16, CIN / 16, Bn), 256, 0, stream>>>(x, xT);
    k_zero_halo<<<(Bn * 2 * CIN + 255) / 256, 256, 0, stream>>>(xT);
    k_conv_wmma<<<Bn * MTILES * 4, 256, 0, stream>>>(xT, Aw, bias, qkv);          // 832 blocks, 8 waves ea.
    k_softmax<<<Bn * Kn, 256, 0, stream>>>(qkv, smA);
    k_lambda<<<Bn, 256, 0, stream>>>(smA, qkv, lam);
    k_qprep<<<dim3(Ln / 16, Bn), 256, 0, stream>>>(qkv, emb, qT, sArr);
    k_output<<<(Bn * HEADSn * 8) / 8, 256, 0, stream>>>(qT, lam, sArr, qkv, out); // 64 blocks, 8 waves ea.
}
